// HSTv3Ultra_84963043049759
// MI455X (gfx1250) — matrix operand order repaired and checked by reading, not compiled, not run on hardware
//
#include <hip/hip_runtime.h>

#define D_MODEL 1024
#define N_HEADS 16
#define HEAD_DIM 64
#define D_FF    4096
#define BATCH   4
#define SEQ     2048
#define NROWS   (BATCH * SEQ)   // 8192

typedef __bf16 bf16_t;
typedef __attribute__((ext_vector_type(16))) __bf16 v16bf;
typedef __attribute__((ext_vector_type(8)))  __bf16 v8bf;
typedef __attribute__((ext_vector_type(8)))  float  v8f;

#define CONCAT16(lo, hi) __builtin_shufflevector((lo), (hi), 0, 1, 2, 3, 4, 5, 6, 7, 8, 9, 10, 11, 12, 13, 14, 15)

__device__ __forceinline__ bf16_t f2bf(float f) {
    unsigned u = __builtin_bit_cast(unsigned, f);
    unsigned r = u + 0x7FFFu + ((u >> 16) & 1u);
    unsigned short h = (unsigned short)(r >> 16);
    return __builtin_bit_cast(bf16_t, h);
}

// ---------------------------------------------------------------------------
// block-wide sum reduction (blockDim.x == 256 -> 8 waves of 32)
// ---------------------------------------------------------------------------
__device__ __forceinline__ float block_sum(float v, float* sm) {
#pragma unroll
    for (int off = 16; off; off >>= 1) v += __shfl_xor(v, off, 32);
    const int lane = threadIdx.x & 31, w = threadIdx.x >> 5;
    if (lane == 0) sm[w] = v;
    __syncthreads();
    if (w == 0) {
        float r = (lane < 8) ? sm[lane] : 0.f;
#pragma unroll
        for (int off = 4; off; off >>= 1) r += __shfl_xor(r, off, 32);
        if (lane == 0) sm[0] = r;
    }
    __syncthreads();
    float r = sm[0];
    __syncthreads();
    return r;
}

// ---------------------------------------------------------------------------
// fp32 [K,N] -> bf16 [N,K]  (transpose + convert), dims multiples of 32
// grid: (K/32, N/32), block: (32,8)
// ---------------------------------------------------------------------------
__global__ __launch_bounds__(256)
void transpose_bf16(const float* __restrict__ W, bf16_t* __restrict__ Wt, int K, int N) {
    __shared__ float tile[32][33];
    const int kb = blockIdx.x * 32, nb = blockIdx.y * 32;
    const int tx = threadIdx.x, ty = threadIdx.y;
#pragma unroll
    for (int i = 0; i < 32; i += 8)
        tile[ty + i][tx] = W[(size_t)(kb + ty + i) * N + nb + tx];
    __syncthreads();
#pragma unroll
    for (int i = 0; i < 32; i += 8)
        Wt[(size_t)(nb + ty + i) * K + kb + tx] = f2bf(tile[tx][ty + i]);
}

// ---------------------------------------------------------------------------
// LayerNorm over D=1024, one block (256 thr) per row, bf16 output
// ---------------------------------------------------------------------------
__global__ __launch_bounds__(256)
void layernorm_bf16(const float* __restrict__ x, const float* __restrict__ g,
                    const float* __restrict__ beta, bf16_t* __restrict__ out) {
    __shared__ float sred[8];
    const int row = blockIdx.x;
    const float* xr = x + (size_t)row * D_MODEL;
    float v[4];
    float s = 0.f;
#pragma unroll
    for (int i = 0; i < 4; i++) { v[i] = xr[threadIdx.x + i * 256]; s += v[i]; }
    s = block_sum(s, sred);
    const float mean = s * (1.f / D_MODEL);
    float q = 0.f;
#pragma unroll
    for (int i = 0; i < 4; i++) { float d = v[i] - mean; q += d * d; }
    q = block_sum(q, sred);
    const float inv = rsqrtf(q * (1.f / D_MODEL) + 1e-5f);
#pragma unroll
    for (int i = 0; i < 4; i++) {
        const int idx = threadIdx.x + i * 256;
        out[(size_t)row * D_MODEL + idx] = f2bf((v[i] - mean) * inv * g[idx] + beta[idx]);
    }
}

// ---------------------------------------------------------------------------
// WMMA GEMM:  C[M,N] = A[M,K](bf16,row) @ Bt[N,K](bf16,row, i.e. B transposed)
// block 256 thr = 8 waves (4m x 2n), block tile 128x128, wave tile 32x64.
// B tile (128 rows x 32 K) staged in LDS via GLOBAL_LOAD_ASYNC_TO_LDS_B128,
// double-buffered (ASYNCcnt + s_wait_asynccnt), 80B row stride (bank-friendly,
// 16B aligned for ds_load_b128).
// MODE 1: out bf16 = relu(acc + bias[n])
// MODE 2: out f32  = acc + (bias?bias[n]:0) + resid[m*N+n]
// MODE 3: out bf16 scattered to [B,H,S,64]   (q/k)
// MODE 4: out bf16 scattered to [B,H,64,S]   (v transposed)
// ---------------------------------------------------------------------------
template <int MODE>
__global__ __launch_bounds__(256)
void gemm_wmma(const bf16_t* __restrict__ A, const bf16_t* __restrict__ Bt,
               const float* __restrict__ bias, const float* __restrict__ resid,
               void* __restrict__ outp, int M, int N, int K) {
    __shared__ __align__(16) bf16_t ldsB[2][128 * 40];  // 2 x 10KB, row stride 40 elems (80B)
    const int tid = threadIdx.x;
    const int lane = tid & 31;
    const int wave = tid >> 5;
    const int wm = wave >> 1, wn = wave & 1;
    const int half = lane >> 4, r = lane & 15;
    const int m_base = blockIdx.x * 128 + wm * 32;
    const int n0 = blockIdx.y * 128;
    const int n_base = n0 + wn * 64;

    // async-copy work split: 512 16B chunks per 128x32 tile -> 2 per thread
    const int c0_row = tid >> 2, c0_col = tid & 3;
    const int c1_row = 64 + (tid >> 2), c1_col = tid & 3;

    v8f c[2][4] = {};

    const bf16_t* a0p = A + (size_t)(m_base + r) * K;
    const bf16_t* a1p = A + (size_t)(m_base + 16 + r) * K;

    const int nk = K >> 5;

    // prologue: stream tile 0 into buffer 0
    {
        const bf16_t* g0 = Bt + (size_t)(n0 + c0_row) * K + c0_col * 8;
        const bf16_t* g1 = Bt + (size_t)(n0 + c1_row) * K + c1_col * 8;
        unsigned l0 = (unsigned)(size_t)&ldsB[0][c0_row * 40 + c0_col * 8];
        unsigned l1 = (unsigned)(size_t)&ldsB[0][c1_row * 40 + c1_col * 8];
        asm volatile("global_load_async_to_lds_b128 %0, %1, off"
                     :: "v"(l0), "v"((unsigned long long)(size_t)g0) : "memory");
        asm volatile("global_load_async_to_lds_b128 %0, %1, off"
                     :: "v"(l1), "v"((unsigned long long)(size_t)g1) : "memory");
    }

    for (int i = 0; i < nk; ++i) {
        const int kt = i << 5;
        // my async loads for tile i are done ...
        asm volatile("s_wait_asynccnt 0x0" ::: "memory");
        // ... and everyone's are; also guarantees buffer (i+1)&1 fully consumed
        __syncthreads();
        if (i + 1 < nk) {
            const int buf = (i + 1) & 1;
            const int ktn = kt + 32;
            const bf16_t* g0 = Bt + (size_t)(n0 + c0_row) * K + ktn + c0_col * 8;
            const bf16_t* g1 = Bt + (size_t)(n0 + c1_row) * K + ktn + c1_col * 8;
            unsigned l0 = (unsigned)(size_t)&ldsB[buf][c0_row * 40 + c0_col * 8];
            unsigned l1 = (unsigned)(size_t)&ldsB[buf][c1_row * 40 + c1_col * 8];
            asm volatile("global_load_async_to_lds_b128 %0, %1, off"
                         :: "v"(l0), "v"((unsigned long long)(size_t)g0) : "memory");
            asm volatile("global_load_async_to_lds_b128 %0, %1, off"
                         :: "v"(l1), "v"((unsigned long long)(size_t)g1) : "memory");
        }

        v16bf a0, a1;
        {
            v8bf lo = *(const v8bf*)(a0p + kt + half * 8);
            v8bf hi = *(const v8bf*)(a0p + kt + 16 + half * 8);
            a0 = CONCAT16(lo, hi);
        }
        {
            v8bf lo = *(const v8bf*)(a1p + kt + half * 8);
            v8bf hi = *(const v8bf*)(a1p + kt + 16 + half * 8);
            a1 = CONCAT16(lo, hi);
        }

        const int buf = i & 1;
#pragma unroll
        for (int nt = 0; nt < 4; ++nt) {
            v16bf b = CONCAT16(*(const v8bf*)&ldsB[buf][(wn * 64 + nt * 16 + r) * 40 + half * 8], *(const v8bf*)&ldsB[buf][(wn * 64 + nt * 16 + r) * 40 + 16 + half * 8]);
            c[0][nt] = __builtin_amdgcn_wmma_f32_16x16x32_bf16(false, a0, false, b, (short)0, c[0][nt], false, false);
            c[1][nt] = __builtin_amdgcn_wmma_f32_16x16x32_bf16(false, a1, false, b, (short)0, c[1][nt], false, false);
        }
    }

#pragma unroll
    for (int mt = 0; mt < 2; ++mt) {
#pragma unroll
        for (int nt = 0; nt < 4; ++nt) {
#pragma unroll
            for (int v = 0; v < 8; ++v) {
                const int m = m_base + mt * 16 + half * 8 + v;
                const int n = n_base + nt * 16 + r;
                const float acc = c[mt][nt][v];
                if (MODE == 1) {
                    float val = acc + bias[n];
                    val = val > 0.f ? val : 0.f;
                    ((bf16_t*)outp)[(size_t)m * N + n] = f2bf(val);
                } else if (MODE == 2) {
                    float val = acc + (bias ? bias[n] : 0.f) + resid[(size_t)m * N + n];
                    ((float*)outp)[(size_t)m * N + n] = val;
                } else if (MODE == 3) {
                    const int b = m >> 11, s = m & 2047, hh = n >> 6, d = n & 63;
                    ((bf16_t*)outp)[((((size_t)b * N_HEADS + hh) * SEQ + s) << 6) + d] = f2bf(acc);
                } else if (MODE == 4) {
                    const int b = m >> 11, s = m & 2047, hh = n >> 6, d = n & 63;
                    ((bf16_t*)outp)[(((size_t)b * N_HEADS + hh) * 64 + d) * SEQ + s] = f2bf(acc);
                }
            }
        }
    }
}

// ---------------------------------------------------------------------------
// Flash attention, causal. One wave per (b,h,32-row q block).
// Q,K: [B,H,S,64] bf16 ; Vt: [B,H,64,S] bf16 ; ctx: [B,S,D] bf16
// ---------------------------------------------------------------------------
__global__ __launch_bounds__(128)
void attn_wmma(const bf16_t* __restrict__ Q, const bf16_t* __restrict__ Km,
               const bf16_t* __restrict__ Vt, bf16_t* __restrict__ ctx) {
    __shared__ __align__(16) bf16_t ldsP[4][32][48];  // per-wave 32x32 P tile (padded)
    const int lane = threadIdx.x & 31;
    const int wave = threadIdx.x >> 5;
    const int gid = blockIdx.x * 4 + wave;            // 0..4095
    const int qblk = gid & 63;
    const int h = (gid >> 6) & 15;
    const int b = gid >> 10;
    const int half = lane >> 4, r = lane & 15;
    const int q0 = qblk * 32;
    const size_t bh = (size_t)b * N_HEADS + h;
    const bf16_t* qp = Q + bh * SEQ * 64;
    const bf16_t* kp = Km + bh * SEQ * 64;
    const bf16_t* vp = Vt + bh * (size_t)64 * SEQ;

    // Q fragments (rows q0..q0+31, K-dim 64 split in two 32-chunks)
    v16bf qa[2][2];
#pragma unroll
    for (int mt = 0; mt < 2; ++mt) {
        const size_t row = (size_t)(q0 + mt * 16 + r) * 64;
#pragma unroll
        for (int kc = 0; kc < 2; ++kc) {
            v8bf lo = *(const v8bf*)(qp + row + kc * 32 + half * 8);
            v8bf hi = *(const v8bf*)(qp + row + kc * 32 + 16 + half * 8);
            qa[mt][kc] = CONCAT16(lo, hi);
        }
    }

    float rm[2][8], rl[2][8];
    v8f o[2][4];
#pragma unroll
    for (int mt = 0; mt < 2; ++mt) {
#pragma unroll
        for (int v = 0; v < 8; ++v) { rm[mt][v] = -3.0e38f; rl[mt][v] = 0.f; }
#pragma unroll
        for (int nt = 0; nt < 4; ++nt) o[mt][nt] = {};
    }

    for (int kb = 0; kb <= qblk; ++kb) {
        const int key0 = kb * 32;
        v16bf kf[2][2];
#pragma unroll
        for (int nt = 0; nt < 2; ++nt)
#pragma unroll
            for (int kc = 0; kc < 2; ++kc)
                kf[nt][kc] = CONCAT16(*(const v8bf*)(kp + (size_t)(key0 + nt * 16 + r) * 64 + kc * 32 + half * 8), *(const v8bf*)(kp + (size_t)(key0 + nt * 16 + r) * 64 + kc * 32 + 16 + half * 8));

        v8f s00 = {}, s01 = {}, s10 = {}, s11 = {};
        s00 = __builtin_amdgcn_wmma_f32_16x16x32_bf16(false, qa[0][0], false, kf[0][0], (short)0, s00, false, false);
        s00 = __builtin_amdgcn_wmma_f32_16x16x32_bf16(false, qa[0][1], false, kf[0][1], (short)0, s00, false, false);
        s01 = __builtin_amdgcn_wmma_f32_16x16x32_bf16(false, qa[0][0], false, kf[1][0], (short)0, s01, false, false);
        s01 = __builtin_amdgcn_wmma_f32_16x16x32_bf16(false, qa[0][1], false, kf[1][1], (short)0, s01, false, false);
        s10 = __builtin_amdgcn_wmma_f32_16x16x32_bf16(false, qa[1][0], false, kf[0][0], (short)0, s10, false, false);
        s10 = __builtin_amdgcn_wmma_f32_16x16x32_bf16(false, qa[1][1], false, kf[0][1], (short)0, s10, false, false);
        s11 = __builtin_amdgcn_wmma_f32_16x16x32_bf16(false, qa[1][0], false, kf[1][0], (short)0, s11, false, false);
        s11 = __builtin_amdgcn_wmma_f32_16x16x32_bf16(false, qa[1][1], false, kf[1][1], (short)0, s11, false, false);

        const bool diag = (kb == qblk);
#pragma unroll
        for (int mt = 0; mt < 2; ++mt) {
            v8f sv0 = mt ? s10 : s00;
            v8f sv1 = mt ? s11 : s01;
#pragma unroll
            for (int v = 0; v < 8; ++v) {
                const int mg = q0 + mt * 16 + half * 8 + v;
                float x0 = sv0[v] * 0.125f;  // 1/sqrt(64)
                float x1 = sv1[v] * 0.125f;
                if (diag) {
                    if (key0 + r > mg) x0 = -1e30f;
                    if (key0 + 16 + r > mg) x1 = -1e30f;
                }
                float mx = fmaxf(x0, x1);
#pragma unroll
                for (int off = 8; off; off >>= 1) mx = fmaxf(mx, __shfl_xor(mx, off, 32));
                const float mold = rm[mt][v];
                const float mnew = fmaxf(mold, mx);
                const float alpha = __expf(mold - mnew);
                const float p0 = __expf(x0 - mnew);
                const float p1 = __expf(x1 - mnew);
                float rs = p0 + p1;
#pragma unroll
                for (int off = 8; off; off >>= 1) rs += __shfl_xor(rs, off, 32);
                rl[mt][v] = rl[mt][v] * alpha + rs;
                rm[mt][v] = mnew;
#pragma unroll
                for (int nt = 0; nt < 4; ++nt) o[mt][nt][v] *= alpha;
                ldsP[wave][mt * 16 + half * 8 + v][r] = f2bf(p0);
                ldsP[wave][mt * 16 + half * 8 + v][16 + r] = f2bf(p1);
            }
        }
        // wave-internal LDS RAW: DS ops are in-order per wave; wait them out
        asm volatile("s_wait_dscnt 0x0" ::: "memory");

        // P A-fragments from LDS (16x32 per mt)
        v16bf pa[2];
#pragma unroll
        for (int mt = 0; mt < 2; ++mt) {
            v8bf lo = *(const v8bf*)&ldsP[wave][mt * 16 + r][half * 8];
            v8bf hi = *(const v8bf*)&ldsP[wave][mt * 16 + r][16 + half * 8];
            pa[mt] = CONCAT16(lo, hi);
        }
        // V B-fragments (Vt rows = head dim, contiguous along keys)
#pragma unroll
        for (int nt = 0; nt < 4; ++nt) {
            v16bf vf = CONCAT16(*(const v8bf*)(vp + (size_t)(nt * 16 + r) * SEQ + key0 + half * 8), *(const v8bf*)(vp + (size_t)(nt * 16 + r) * SEQ + key0 + 16 + half * 8));
            o[0][nt] = __builtin_amdgcn_wmma_f32_16x16x32_bf16(false, pa[0], false, vf, (short)0, o[0][nt], false, false);
            o[1][nt] = __builtin_amdgcn_wmma_f32_16x16x32_bf16(false, pa[1], false, vf, (short)0, o[1][nt], false, false);
        }
    }

    // normalize and write ctx [B,S,D] bf16
#pragma unroll
    for (int mt = 0; mt < 2; ++mt) {
#pragma unroll
        for (int nt = 0; nt < 4; ++nt) {
#pragma unroll
            for (int v = 0; v < 8; ++v) {
                const float val = o[mt][nt][v] / rl[mt][v];
                const int s = q0 + mt * 16 + half * 8 + v;
                const int d = h * 64 + nt * 16 + r;
                ctx[((size_t)b * SEQ + s) * D_MODEL + d] = f2bf(val);
            }
        }
    }
}

// ---------------------------------------------------------------------------
// pooled[b,d] = mean_s x_out[b,s,d]       (4096 threads, coalesced over d)
// ---------------------------------------------------------------------------
__global__ __launch_bounds__(256)
void pool_kernel(const float* __restrict__ xo, float* __restrict__ pooled) {
    const int t = blockIdx.x * 256 + threadIdx.x;  // 0..4095
    const int b = t >> 10, d = t & 1023;
    const float* p = xo + (size_t)b * SEQ * D_MODEL + d;
    float s = 0.f;
    for (int i = 0; i < SEQ; ++i) s += p[(size_t)i * D_MODEL];
    pooled[t] = s * (1.f / SEQ);
}

// conf = mean_b sigmoid(pooled[b] . Wc + bc)  -> out[0]
__global__ __launch_bounds__(256)
void conf_kernel(const float* __restrict__ pooled, const float* __restrict__ Wc,
                 const float* __restrict__ bc, float* __restrict__ out) {
    __shared__ float sred[8];
    float acc[4] = {0.f, 0.f, 0.f, 0.f};
    for (int d = threadIdx.x; d < D_MODEL; d += 256) {
        const float w = Wc[d];
#pragma unroll
        for (int b = 0; b < 4; ++b) acc[b] += pooled[b * D_MODEL + d] * w;
    }
    float tot[4];
#pragma unroll
    for (int b = 0; b < 4; ++b) tot[b] = block_sum(acc[b], sred);
    if (threadIdx.x == 0) {
        float s = 0.f;
#pragma unroll
        for (int b = 0; b < 4; ++b) {
            const float z = tot[b] + bc[0];
            s += 1.f / (1.f + __expf(-z));
        }
        out[0] = s * 0.25f;
    }
}

// ---------------------------------------------------------------------------
extern "C" void kernel_launch(void* const* d_in, const int* in_sizes, int n_in,
                              void* d_out, int out_size, void* d_ws, size_t ws_size,
                              hipStream_t stream) {
    const float* x     = (const float*)d_in[0];
    const float* Wq    = (const float*)d_in[1];
    const float* Wk    = (const float*)d_in[2];
    const float* Wv    = (const float*)d_in[3];
    const float* Wo    = (const float*)d_in[4];
    const float* ln1_g = (const float*)d_in[5];
    const float* ln1_b = (const float*)d_in[6];
    const float* ln2_g = (const float*)d_in[7];
    const float* ln2_b = (const float*)d_in[8];
    const float* W1    = (const float*)d_in[9];
    const float* b1    = (const float*)d_in[10];
    const float* W2    = (const float*)d_in[11];
    const float* b2    = (const float*)d_in[12];
    const float* Wc    = (const float*)d_in[13];
    const float* bc    = (const float*)d_in[14];

    // workspace layout (bf16 elements unless noted)
    bf16_t* Wqt = (bf16_t*)d_ws;                         // [1024,1024]
    bf16_t* Wkt = Wqt + 1024 * 1024;
    bf16_t* Wvt = Wkt + 1024 * 1024;
    bf16_t* Wvt2 = Wvt;                                  // alias clarity
    bf16_t* Wot = Wvt + 1024 * 1024;
    bf16_t* W1t = Wot + 1024 * 1024;                     // [4096,1024]
    bf16_t* W2t = W1t + (size_t)4096 * 1024;             // [1024,4096]
    bf16_t* hbuf = W2t + (size_t)4096 * 1024;            // [8192,1024]
    bf16_t* qbuf = hbuf + (size_t)NROWS * D_MODEL;       // [B,H,S,64]
    bf16_t* kbuf = qbuf + (size_t)NROWS * D_MODEL;
    bf16_t* vbuf = kbuf + (size_t)NROWS * D_MODEL;
    bf16_t* ctxb = vbuf + (size_t)NROWS * D_MODEL;       // [8192,1024]
    bf16_t* ffb  = qbuf;                                  // reuse q..ctx = [8192,4096]
    float*  x1   = (float*)(ctxb + (size_t)NROWS * D_MODEL);
    float*  pooled = x1 + (size_t)NROWS * D_MODEL;
    (void)Wvt2;

    float* xout = (float*)d_out;                          // [8192,1024] + conf at end

    const dim3 tb(32, 8);
    transpose_bf16<<<dim3(32, 32), tb, 0, stream>>>(Wq, Wqt, 1024, 1024);
    transpose_bf16<<<dim3(32, 32), tb, 0, stream>>>(Wk, Wkt, 1024, 1024);
    transpose_bf16<<<dim3(32, 32), tb, 0, stream>>>(Wv, Wvt, 1024, 1024);
    transpose_bf16<<<dim3(32, 32), tb, 0, stream>>>(Wo, Wot, 1024, 1024);
    transpose_bf16<<<dim3(32, 128), tb, 0, stream>>>(W1, W1t, 1024, 4096);
    transpose_bf16<<<dim3(128, 32), tb, 0, stream>>>(W2, W2t, 4096, 1024);

    layernorm_bf16<<<NROWS, 256, 0, stream>>>(x, ln1_g, ln1_b, hbuf);

    gemm_wmma<3><<<dim3(64, 8), 256, 0, stream>>>(hbuf, Wqt, nullptr, nullptr, qbuf, NROWS, 1024, 1024);
    gemm_wmma<3><<<dim3(64, 8), 256, 0, stream>>>(hbuf, Wkt, nullptr, nullptr, kbuf, NROWS, 1024, 1024);
    gemm_wmma<4><<<dim3(64, 8), 256, 0, stream>>>(hbuf, Wvt, nullptr, nullptr, vbuf, NROWS, 1024, 1024);

    attn_wmma<<<1024, 128, 0, stream>>>(qbuf, kbuf, vbuf, ctxb);

    gemm_wmma<2><<<dim3(64, 8), 256, 0, stream>>>(ctxb, Wot, nullptr, x, x1, NROWS, 1024, 1024);

    layernorm_bf16<<<NROWS, 256, 0, stream>>>(x1, ln2_g, ln2_b, hbuf);

    gemm_wmma<1><<<dim3(64, 32), 256, 0, stream>>>(hbuf, W1t, b1, nullptr, ffb, NROWS, D_FF, 1024);
    gemm_wmma<2><<<dim3(64, 8), 256, 0, stream>>>(ffb, W2t, b2, x1, xout, NROWS, 1024, D_FF);

    pool_kernel<<<16, 256, 0, stream>>>(xout, pooled);
    conf_kernel<<<1, 256, 0, stream>>>(pooled, Wc, bc, xout + (size_t)NROWS * D_MODEL);
}
